// E28Cell_2147483648047
// MI455X (gfx1250) — matrix operand order repaired and checked by reading, not compiled, not run on hardware
//
#include <hip/hip_runtime.h>
#include <math.h>

// ---------------------------------------------------------------------------
// E28Cell on gfx1250 (MI455X): conv+silu -> bf16 WMMA GEMM (xw = xc*Wx^T + b)
// -> grid-synchronized sequential Elman recurrence with W_h slices in LDS.
// Round 3: enforce the 2-deep software pipeline with
// __builtin_amdgcn_sched_barrier(0) stage fences so the MI scheduler cannot
// sink next-stage fragment loads below the current WMMAs (which previously
// recreated s_wait_loadcnt 0 serialization via VGPR reuse).
// ---------------------------------------------------------------------------

typedef __attribute__((ext_vector_type(16))) __bf16   v16bf;
typedef __attribute__((ext_vector_type(8)))  float    v8f;
typedef __attribute__((ext_vector_type(4)))  unsigned uv4;

#define DIM   1280
#define BB    4
#define TT    2048
#define BT    (BB * TT)
#define KCONV 4

#define SCHED_FENCE() __builtin_amdgcn_sched_barrier(0)

union FragBF { uv4 q[2]; v16bf v; };   // 32 bytes: one A or B WMMA operand

// A-frag (16x32 bf16): lane holds 16 contiguous elems at p
__device__ __forceinline__ void load_a(FragBF& f, const unsigned short* p) {
  f.q[0] = *(const uv4*)(p);
  f.q[1] = *(const uv4*)(p + 8);
}
// B-frag (32x16 bf16): lane holds 16 contiguous elems at p
__device__ __forceinline__ void load_b(FragBF& f, const unsigned short* p) {
  f.q[0] = *(const uv4*)(p);
  f.q[1] = *(const uv4*)(p + 8);
}

__device__ __forceinline__ unsigned short f2bf(float f) {
  union { float f; unsigned u; } cv; cv.f = f;
  unsigned u = cv.u;
  u += 0x7FFFu + ((u >> 16) & 1u);          // round-to-nearest-even
  return (unsigned short)(u >> 16);
}

__device__ __forceinline__ float sigm(float x) { return 1.0f / (1.0f + __expf(-x)); }

// ---------------------------------------------------------------------------
// Kernel 1: cast W_x and W_h to bf16 (row-major [e][d], e-rows contiguous)
// ---------------------------------------------------------------------------
__global__ void k_cast_weights(const float* __restrict__ Wx,
                               const float* __restrict__ Wh,
                               unsigned short* __restrict__ Wxb,
                               unsigned short* __restrict__ Whb, int n) {
  int i = blockIdx.x * blockDim.x + threadIdx.x;
  if (i < n) { Wxb[i] = f2bf(Wx[i]); Whb[i] = f2bf(Wh[i]); }
}

// ---------------------------------------------------------------------------
// Kernel 2: causal depthwise conv (K=4) + bias + SiLU, output bf16 [BT][DIM]
// ---------------------------------------------------------------------------
__global__ void k_conv_silu(const float* __restrict__ x,
                            const float* __restrict__ cw,
                            const float* __restrict__ cb,
                            unsigned short* __restrict__ xc) {
  int i = blockIdx.x * blockDim.x + threadIdx.x;   // i = (b*TT+t)*DIM + d
  if (i >= BT * DIM) return;
  int d = i % DIM;
  int m = i / DIM;
  int t = m % TT;
  float acc = cb[d];
#pragma unroll
  for (int k = 0; k < KCONV; ++k) {
    int tt = t - (KCONV - 1) + k;
    if (tt >= 0) acc += cw[d * KCONV + k] * x[i + (k - (KCONV - 1)) * DIM];
  }
  float s = acc * sigm(acc);
  xc[i] = f2bf(s);
}

// ---------------------------------------------------------------------------
// Kernel 3: init padded hidden-state ping-pong buffers (16 x DIM bf16) + ctr
// rows 0..3 of buf0 = h0; rows 4..15 of both buffers stay zero forever.
// ---------------------------------------------------------------------------
__global__ void k_init(const float* __restrict__ h0,
                       unsigned short* __restrict__ hb0,
                       unsigned short* __restrict__ hb1,
                       unsigned* __restrict__ cnt) {
  int i = blockIdx.x * blockDim.x + threadIdx.x;
  if (i == 0) *cnt = 0u;
  if (i < 16 * DIM) {
    int row = i / DIM, col = i % DIM;
    hb0[i] = (row < BB) ? f2bf(h0[row * DIM + col]) : (unsigned short)0;
    hb1[i] = (unsigned short)0;
  }
}

// ---------------------------------------------------------------------------
// Kernel 4: xw = xc * Wx^T + b   (bf16 WMMA, f32 accumulate)
// Each wave computes a 16x64 tile; 2-deep pipelined k-loop with stage fences.
// Grid: 512 M-tiles * 20 N-groups = 10240 waves = 1280 blocks * 8 waves.
// ---------------------------------------------------------------------------
__global__ void __launch_bounds__(256)
k_gemm1(const unsigned short* __restrict__ xc,
        const unsigned short* __restrict__ Wxb,
        const float* __restrict__ bias,
        float* __restrict__ xw) {
  int lane = threadIdx.x & 31;
  int wave = blockIdx.x * (blockDim.x >> 5) + (threadIdx.x >> 5);
  int mtile = wave / 20;
  int ng    = wave % 20;
  int m0 = mtile * 16;
  int e0 = ng * 64;
  int lm = lane & 15;
  int lh = lane >> 4;

  const unsigned short* arow = xc + (size_t)(m0 + lm) * DIM + 16 * lh;
  const unsigned short* brow = Wxb + (size_t)(e0 + lm) * DIM + 16 * lh;

  v8f c[4] = {v8f{}, v8f{}, v8f{}, v8f{}};

  FragBF a0, a1, b0[4], b1[4];
  // prologue: fragments for kk = 0
  load_a(a0, arow);
#pragma unroll
  for (int j = 0; j < 4; ++j) load_b(b0[j], brow + (size_t)j * 16 * DIM);
  SCHED_FENCE();

  for (int kk = 0; kk < DIM; kk += 64) {
    // stage: loads for kk+32 (always valid: last kk=1216 -> 1248 < 1280)
    load_a(a1, arow + kk + 32);
#pragma unroll
    for (int j = 0; j < 4; ++j) load_b(b1[j], brow + (size_t)j * 16 * DIM + kk + 32);
    SCHED_FENCE();
    // stage: compute kk (operands loaded one stage earlier)
#pragma unroll
    for (int j = 0; j < 4; ++j)
      c[j] = __builtin_amdgcn_wmma_f32_16x16x32_bf16(
          false, a0.v, false, b0[j].v, (short)0, c[j], false, false);
    SCHED_FENCE();
    // stage: loads for kk+64 (skip refill past the end)
    if (kk + 64 < DIM) {
      load_a(a0, arow + kk + 64);
#pragma unroll
      for (int j = 0; j < 4; ++j) load_b(b0[j], brow + (size_t)j * 16 * DIM + kk + 64);
    }
    SCHED_FENCE();
    // stage: compute kk+32
#pragma unroll
    for (int j = 0; j < 4; ++j)
      c[j] = __builtin_amdgcn_wmma_f32_16x16x32_bf16(
          false, a1.v, false, b1[j].v, (short)0, c[j], false, false);
    SCHED_FENCE();
  }

#pragma unroll
  for (int j = 0; j < 4; ++j) {
    int col = e0 + j * 16 + lm;
    float bv = bias[col];
#pragma unroll
    for (int r = 0; r < 8; ++r) {
      int m = m0 + r + 8 * lh;                  // C/D: M = r + 8*(lane/16)
      xw[(size_t)m * DIM + col] = c[j][r] + bv;
    }
  }
}

// ---------------------------------------------------------------------------
// Kernel 5: sequential recurrence h_t = tanh(xw_t + h_{t-1} W_h^T),
// out_t = h_t * silu(z_t). 80 single-wave blocks, each owns 16 e-columns and
// keeps its 1280x16 slice of W_h in LDS (40 KB). Device-scope atomic grid
// barrier chains the 2048 steps; hidden state ping-pongs through L2 as bf16
// padded to 16 rows (rows 4..15 zero) so it loads directly as a WMMA A-frag.
// k-loop: 2-deep pipeline with stage fences (global A + LDS B loads for the
// next k-step issue under the current WMMA).
// ---------------------------------------------------------------------------
__global__ void __launch_bounds__(32)
k_rnn(const unsigned short* __restrict__ Whb,
      const float* __restrict__ xw,
      const float* __restrict__ z,
      unsigned short* __restrict__ hb0,
      unsigned short* __restrict__ hb1,
      float* __restrict__ out,
      float* __restrict__ hlast,
      unsigned* __restrict__ cnt) {
  __shared__ unsigned short ldsW[16 * DIM];          // 40 KB W_h slice
  int lane = threadIdx.x;
  int e0 = blockIdx.x * 16;

  // Stage 16 consecutive W_h rows (contiguous 40 KB) into LDS.
  {
    const uv4* src = (const uv4*)(Whb + (size_t)e0 * DIM);
    uv4* dst = (uv4*)ldsW;
    for (int i = lane; i < (16 * DIM) / 8; i += 32) dst[i] = src[i];
  }
  __syncthreads();

  int lm = lane & 15, lh = lane >> 4;
  const unsigned short* ldsB = ldsW + lm * DIM + 16 * lh;  // B-frag base in LDS
  int aoff = lm * DIM + 16 * lh;                           // A-frag base offset
  int col = e0 + lm;
  unsigned nblk = gridDim.x;

  for (int t = 0; t < TT; ++t) {
    const unsigned short* hin  = (t & 1) ? hb1 : hb0;
    unsigned short*       hout = (t & 1) ? hb0 : hb1;

    // Accumulator seeded with xw (bias already folded in): D = h*Wh^T + xw
    v8f c = {};
    if (lane < 16) {
#pragma unroll
      for (int r = 0; r < BB; ++r)                    // M = r (lh==0, r<4)
        c[r] = xw[(size_t)(r * TT + t) * DIM + col];
    }

    FragBF a0, a1, b0, b1;
    load_a(a0, hin + aoff);
    load_b(b0, ldsB);
    SCHED_FENCE();
    for (int kk = 0; kk < DIM; kk += 64) {
      load_a(a1, hin + aoff + kk + 32);               // 1248 < 1280: always ok
      load_b(b1, ldsB + kk + 32);
      SCHED_FENCE();
      c = __builtin_amdgcn_wmma_f32_16x16x32_bf16(
          false, a0.v, false, b0.v, (short)0, c, false, false);
      SCHED_FENCE();
      if (kk + 64 < DIM) {
        load_a(a0, hin + aoff + kk + 64);
        load_b(b0, ldsB + kk + 64);
      }
      SCHED_FENCE();
      c = __builtin_amdgcn_wmma_f32_16x16x32_bf16(
          false, a1.v, false, b1.v, (short)0, c, false, false);
      SCHED_FENCE();
    }

    if (lane < 16) {
#pragma unroll
      for (int r = 0; r < BB; ++r) {
        float h = tanhf(c[r]);
        size_t idx = (size_t)(r * TT + t) * DIM + col;
        float zv = z[idx];
        out[idx] = h * (zv * sigm(zv));
        hout[r * DIM + col] = f2bf(h);
        if (t == TT - 1) hlast[r * DIM + col] = h;
      }
      if (t + 1 < TT)                                  // warm L2/L0 for next xw
        __builtin_prefetch(&xw[(size_t)((lane & 3) * TT + t + 1) * DIM + col], 0, 1);
    }

    // ---- device-wide step barrier (release h writes, then rendezvous) ----
    __threadfence();
    if (lane == 0) {
      __hip_atomic_fetch_add(cnt, 1u, __ATOMIC_ACQ_REL, __HIP_MEMORY_SCOPE_AGENT);
      unsigned target = (unsigned)(t + 1) * nblk;
      while (__hip_atomic_load(cnt, __ATOMIC_ACQUIRE, __HIP_MEMORY_SCOPE_AGENT) < target)
        __builtin_amdgcn_s_sleep(1);
    }
    __syncthreads();
    __threadfence();
  }
}

// ---------------------------------------------------------------------------
// Host-side launch
// ---------------------------------------------------------------------------
extern "C" void kernel_launch(void* const* d_in, const int* in_sizes, int n_in,
                              void* d_out, int out_size, void* d_ws, size_t ws_size,
                              hipStream_t stream) {
  (void)in_sizes; (void)n_in; (void)out_size; (void)ws_size;

  const float* x_seq = (const float*)d_in[0];
  const float* z_seq = (const float*)d_in[1];
  const float* h0    = (const float*)d_in[2];
  const float* W_x   = (const float*)d_in[3];
  const float* W_h   = (const float*)d_in[4];
  const float* bvec  = (const float*)d_in[5];
  const float* convw = (const float*)d_in[6];
  const float* convb = (const float*)d_in[7];

  float* out   = (float*)d_out;
  float* hlast = out + (size_t)BT * DIM;

  // Workspace layout (bytes)
  char* ws = (char*)d_ws;
  constexpr size_t XC_B  = (size_t)BT * DIM * 2;      // 20,971,520
  constexpr size_t W_B   = (size_t)DIM * DIM * 2;     //  3,276,800
  constexpr size_t XW_B  = (size_t)BT * DIM * 4;      // 41,943,040
  constexpr size_t HB_B  = (size_t)16 * DIM * 2;      //     40,960
  unsigned short* xc  = (unsigned short*)(ws);
  unsigned short* Wxb = (unsigned short*)(ws + XC_B);
  unsigned short* Whb = (unsigned short*)(ws + XC_B + W_B);
  float*          xw  = (float*)        (ws + XC_B + 2 * W_B);
  unsigned short* hb0 = (unsigned short*)(ws + XC_B + 2 * W_B + XW_B);
  unsigned short* hb1 = (unsigned short*)(ws + XC_B + 2 * W_B + XW_B + HB_B);
  unsigned*       cnt = (unsigned*)      (ws + XC_B + 2 * W_B + XW_B + 2 * HB_B);

  k_cast_weights<<<(DIM * DIM + 255) / 256, 256, 0, stream>>>(W_x, W_h, Wxb, Whb, DIM * DIM);
  k_conv_silu<<<(BT * DIM + 255) / 256, 256, 0, stream>>>(x_seq, convw, convb, xc);
  k_init<<<(16 * DIM + 255) / 256, 256, 0, stream>>>(h0, hb0, hb1, cnt);
  k_gemm1<<<1280, 256, 0, stream>>>(xc, Wxb, bvec, xw);
  k_rnn<<<80, 32, 0, stream>>>(Whb, xw, z_seq, hb0, hb1, out, hlast, cnt);
}
